// AttnDecoder_2972117368952
// MI455X (gfx1250) — matrix operand order repaired and checked by reading, not compiled, not run on hardware
//
#include <hip/hip_runtime.h>
#include <math.h>

typedef __bf16 bf16;
typedef __attribute__((ext_vector_type(16))) __bf16 v16bf;
typedef __attribute__((ext_vector_type(8)))  __bf16 v8bf;
typedef __attribute__((ext_vector_type(8)))  float  v8f;

#define BB 64
#define SS 128
#define TT 32
#define HH 1024
#define VV 32000

// ---------------- f32 -> bf16 conversion (weights, state) ----------------
__global__ __launch_bounds__(256) void k_cvt_bf16(const float* __restrict__ src,
                                                  bf16* __restrict__ dst, size_t n) {
  size_t i = (size_t)blockIdx.x * blockDim.x + threadIdx.x;
  size_t stride = (size_t)gridDim.x * blockDim.x;
  for (; i < n; i += stride) dst[i] = (bf16)src[i];
}

// ---------------- small one-shot f32 NT matmul (bridge, keys) ----------------
// C[m,n] = sum_k A[m,k]*W[n,k] + bias[n]
__global__ __launch_bounds__(256) void k_matmul_nt_f32(const float* __restrict__ A,
                                                       const float* __restrict__ W,
                                                       const float* __restrict__ bias,
                                                       float* __restrict__ C,
                                                       int M, int N, int K) {
  int idx = blockIdx.x * 256 + threadIdx.x;
  if (idx >= M * N) return;
  int m = idx / N, n = idx % N;
  const float4* a = (const float4*)(A + (size_t)m * K);
  const float4* w = (const float4*)(W + (size_t)n * K);
  float s = 0.f;
  for (int k = 0; k < K / 4; ++k) {
    float4 av = a[k], wv = w[k];
    s += av.x * wv.x + av.y * wv.y + av.z * wv.z + av.w * wv.w;
  }
  C[idx] = s + bias[n];
}

// ---------------- bf16 WMMA NT GEMM, M fixed at 64 ----------------
// C[64,N] (+)= A(64xK) * B(NxK)^T + bias[N]
// 256 threads = 8 waves. Each wave owns a 16-wide N strip and ALL 64 M rows:
// 4 accumulators share one B fragment per k-step -> B is loaded exactly once
// globally, 2.5 b128 loads per WMMA instead of 4.
__global__ __launch_bounds__(256) void k_gemm_bf16_nt_m64(const bf16* __restrict__ A,
                                                          const bf16* __restrict__ Bm,
                                                          const float* __restrict__ bias,
                                                          float* __restrict__ C,
                                                          int N, int K,
                                                          long long ldc, int accumulate) {
  const int lane  = threadIdx.x & 31;
  const int wave  = threadIdx.x >> 5;     // 0..7 -> N strip
  const int lrow  = lane & 15;            // M idx (A frag) / N idx (B frag, D col)
  const int lhalf = lane >> 4;            // K-half selector
  const int n0 = blockIdx.x * 128 + wave * 16;
  if (n0 >= N) return;

  const bf16* a0p = A + (size_t)(lrow)      * K;   // M tile 0
  const bf16* a1p = A + (size_t)(16 + lrow) * K;   // M tile 1
  const bf16* a2p = A + (size_t)(32 + lrow) * K;   // M tile 2
  const bf16* a3p = A + (size_t)(48 + lrow) * K;   // M tile 3
  const bf16* brow = Bm + (size_t)(n0 + lrow) * K;

  v8f acc0 = {}, acc1 = {}, acc2 = {}, acc3 = {};
  for (int k = 0; k < K; k += 32) {
    // B 32x16: lanes 0-15 K={0..7,16..23}; lanes 16-31 K={8..15,24..31} (col = lane%16)
    v8bf b0 = *(const v8bf*)(brow + k + lhalf * 8);
    v8bf b1 = *(const v8bf*)(brow + k + 16 + lhalf * 8);
    v16bf bfm;
#pragma unroll
    for (int i = 0; i < 8; ++i) { bfm[i] = b0[i]; bfm[i + 8] = b1[i]; }

    // A 16x32: lanes 0-15 K={0..7,16..23}; lanes 16-31 K={8..15,24..31}
    const int ko0 = k + lhalf * 8;
    const int ko1 = k + 16 + lhalf * 8;
    {
      v8bf x0 = *(const v8bf*)(a0p + ko0), x1 = *(const v8bf*)(a0p + ko1);
      v16bf a;
#pragma unroll
      for (int i = 0; i < 8; ++i) { a[i] = x0[i]; a[i + 8] = x1[i]; }
      acc0 = __builtin_amdgcn_wmma_f32_16x16x32_bf16(false, a, false, bfm, (short)0, acc0, false, false);
    }
    {
      v8bf x0 = *(const v8bf*)(a1p + ko0), x1 = *(const v8bf*)(a1p + ko1);
      v16bf a;
#pragma unroll
      for (int i = 0; i < 8; ++i) { a[i] = x0[i]; a[i + 8] = x1[i]; }
      acc1 = __builtin_amdgcn_wmma_f32_16x16x32_bf16(false, a, false, bfm, (short)0, acc1, false, false);
    }
    {
      v8bf x0 = *(const v8bf*)(a2p + ko0), x1 = *(const v8bf*)(a2p + ko1);
      v16bf a;
#pragma unroll
      for (int i = 0; i < 8; ++i) { a[i] = x0[i]; a[i + 8] = x1[i]; }
      acc2 = __builtin_amdgcn_wmma_f32_16x16x32_bf16(false, a, false, bfm, (short)0, acc2, false, false);
    }
    {
      v8bf x0 = *(const v8bf*)(a3p + ko0), x1 = *(const v8bf*)(a3p + ko1);
      v16bf a;
#pragma unroll
      for (int i = 0; i < 8; ++i) { a[i] = x0[i]; a[i + 8] = x1[i]; }
      acc3 = __builtin_amdgcn_wmma_f32_16x16x32_bf16(false, a, false, bfm, (short)0, acc3, false, false);
    }
  }

  const int col = n0 + lrow;
  const float bv = bias ? bias[col] : 0.f;
  // D layout: VGPR r -> M = r + 8*(lane/16), col = lane%16
#pragma unroll
  for (int r = 0; r < 8; ++r) {
    long long row0 = r + 8 * lhalf;
    float v0 = acc0[r] + bv;
    float v1 = acc1[r] + bv;
    float v2 = acc2[r] + bv;
    float v3 = acc3[r] + bv;
    float* c0 = C + (row0 +  0) * ldc + col;
    float* c1 = C + (row0 + 16) * ldc + col;
    float* c2 = C + (row0 + 32) * ldc + col;
    float* c3 = C + (row0 + 48) * ldc + col;
    if (accumulate) { *c0 += v0; *c1 += v1; *c2 += v2; *c3 += v3; }
    else            { *c0  = v0; *c1  = v1; *c2  = v2; *c3  = v3; }
  }
}

// ---------------- embedding gather into bf16 x[:, 0:H] ----------------
__global__ __launch_bounds__(256) void k_pack_emb(const float* __restrict__ emb,
                                                  const int* __restrict__ tgt,
                                                  bf16* __restrict__ x, int t) {
  int b = blockIdx.x;
  int tok = tgt[b * TT + t];
  const float* e = emb + (size_t)tok * HH;
  for (int h = threadIdx.x; h < HH; h += 256)
    x[(size_t)b * 2 * HH + h] = (bf16)e[h];
}

// ---------------- attention scores: score[b,s] = Vw . tanh(qW1[b]+keys[s]) + Vb ----------------
__global__ __launch_bounds__(256) void k_attn_scores(const float* __restrict__ qW1,
                                                     const float* __restrict__ keys,
                                                     const float* __restrict__ Vw,
                                                     const float* __restrict__ Vb,
                                                     float* __restrict__ scores) {
  int s = blockIdx.x, b = blockIdx.y;
  int h4 = threadIdx.x * 4;  // 256 threads * 4 = H
  const float4 q = *(const float4*)(qW1 + (size_t)b * HH + h4);
  const float4 k = *(const float4*)(keys + (size_t)s * HH + h4);
  const float4 v = *(const float4*)(Vw + h4);
  float p = v.x * tanhf(q.x + k.x) + v.y * tanhf(q.y + k.y) +
            v.z * tanhf(q.z + k.z) + v.w * tanhf(q.w + k.w);
  __shared__ float red[256];
  red[threadIdx.x] = p;
  __syncthreads();
  for (int off = 128; off > 0; off >>= 1) {
    if (threadIdx.x < off) red[threadIdx.x] += red[threadIdx.x + off];
    __syncthreads();
  }
  if (threadIdx.x == 0) scores[b * SS + s] = red[0] + Vb[0];
}

// ---------------- softmax over S, write alphas, context -> bf16 x[:, H:2H] ----------------
__global__ __launch_bounds__(256) void k_softmax_ctx(const float* __restrict__ scores,
                                                     const float* __restrict__ enc,
                                                     float* __restrict__ attn_out,
                                                     bf16* __restrict__ x, int t) {
  int b = blockIdx.x, tid = threadIdx.x;
  __shared__ float sc[SS];
  __shared__ float red[128];
  if (tid < SS) sc[tid] = scores[b * SS + tid];
  __syncthreads();
  if (tid < 64) red[tid] = fmaxf(sc[tid], sc[tid + 64]);
  __syncthreads();
  for (int off = 32; off > 0; off >>= 1) {
    if (tid < off) red[tid] = fmaxf(red[tid], red[tid + off]);
    __syncthreads();
  }
  float mx = red[0];
  __syncthreads();
  if (tid < SS) sc[tid] = expf(sc[tid] - mx);
  __syncthreads();
  if (tid < 64) red[tid] = sc[tid] + sc[tid + 64];
  __syncthreads();
  for (int off = 32; off > 0; off >>= 1) {
    if (tid < off) red[tid] += red[tid + off];
    __syncthreads();
  }
  float inv = 1.f / red[0];
  __syncthreads();
  if (tid < SS) {
    sc[tid] *= inv;
    attn_out[((size_t)t * BB + b) * SS + tid] = sc[tid];  // attn (T,B,S)
  }
  __syncthreads();
  for (int i = 0; i < HH / 256; ++i) {
    int h = tid + i * 256;
    float acc = 0.f;
    const float* ep = enc + (size_t)b * SS * HH + h;
    for (int s2 = 0; s2 < SS; ++s2) acc += sc[s2] * ep[(size_t)s2 * HH];
    x[(size_t)b * 2 * HH + HH + h] = (bf16)acc;
  }
}

// ---------------- LSTM cell elementwise ----------------
__device__ __forceinline__ float sigf(float v) { return 1.f / (1.f + expf(-v)); }

__global__ __launch_bounds__(256) void k_lstm_cell(const float* __restrict__ gates,
                                                   float* __restrict__ hst,
                                                   float* __restrict__ cst,
                                                   bf16* __restrict__ hbf,
                                                   float* __restrict__ out_h,
                                                   float* __restrict__ out_c) {
  int b = blockIdx.x;
  size_t g = (size_t)b * 4 * HH;
  for (int i = 0; i < HH / 256; ++i) {
    int h = threadIdx.x + i * 256;
    float ig = gates[g + h];
    float fg = gates[g + HH + h];
    float gg = gates[g + 2 * HH + h];
    float og = gates[g + 3 * HH + h];
    float cprev = cst[(size_t)b * HH + h];
    float cn = sigf(fg) * cprev + sigf(ig) * tanhf(gg);
    float hn = sigf(og) * tanhf(cn);
    cst[(size_t)b * HH + h] = cn;
    hst[(size_t)b * HH + h] = hn;
    hbf[(size_t)b * HH + h] = (bf16)hn;
    out_h[(size_t)b * HH + h] = hn;   // final step's value survives
    out_c[(size_t)b * HH + h] = cn;
  }
}

// ---------------- in-place log_softmax over V for row (b,t) ----------------
__global__ __launch_bounds__(256) void k_log_softmax(float* __restrict__ logits, int t) {
  int b = blockIdx.x, tid = threadIdx.x;
  float* row = logits + ((size_t)b * TT + t) * (size_t)VV;
  __shared__ float red[256];
  float mx = -1e30f;
  for (int v = tid; v < VV; v += 256) mx = fmaxf(mx, row[v]);
  red[tid] = mx;
  __syncthreads();
  for (int off = 128; off > 0; off >>= 1) {
    if (tid < off) red[tid] = fmaxf(red[tid], red[tid + off]);
    __syncthreads();
  }
  mx = red[0];
  __syncthreads();
  float sm = 0.f;
  for (int v = tid; v < VV; v += 256) sm += expf(row[v] - mx);
  red[tid] = sm;
  __syncthreads();
  for (int off = 128; off > 0; off >>= 1) {
    if (tid < off) red[tid] += red[tid + off];
    __syncthreads();
  }
  float lse = mx + logf(red[0]);
  __syncthreads();
  for (int v = tid; v < VV; v += 256) row[v] -= lse;
}

// ------------------------------------------------------------------------
extern "C" void kernel_launch(void* const* d_in, const int* in_sizes, int n_in,
                              void* d_out, int out_size, void* d_ws, size_t ws_size,
                              hipStream_t stream) {
  const float* enc   = (const float*)d_in[0];   // (B,S,H)
  const float* enc_h = (const float*)d_in[1];   // (1,B,H)
  const float* enc_c = (const float*)d_in[2];   // (1,B,H)
  /* d_in[3] input_mask unused by reference */
  const int*   tgt   = (const int*)d_in[4];     // (B,T)
  const float* emb   = (const float*)d_in[5];   // (V,H)
  const float* W1    = (const float*)d_in[6];
  const float* b1    = (const float*)d_in[7];
  const float* W2    = (const float*)d_in[8];
  const float* b2    = (const float*)d_in[9];
  const float* Vw    = (const float*)d_in[10];
  const float* Vb    = (const float*)d_in[11];
  const float* W_ih  = (const float*)d_in[12];  // (4H,2H)
  const float* W_hh  = (const float*)d_in[13];  // (4H,H)
  const float* b_ih  = (const float*)d_in[14];
  const float* b_hh  = (const float*)d_in[15];
  const float* outW  = (const float*)d_in[16];  // (V,H)
  const float* outb  = (const float*)d_in[17];
  const float* br1W  = (const float*)d_in[18];
  const float* br1b  = (const float*)d_in[19];
  const float* br2W  = (const float*)d_in[20];
  const float* br2b  = (const float*)d_in[21];
  (void)in_sizes; (void)n_in; (void)out_size; (void)ws_size;

  float* out = (float*)d_out;
  const size_t H_OFF = (size_t)BB * TT * VV;           // h (1,B,H)
  const size_t C_OFF = H_OFF + (size_t)BB * HH;        // c (1,B,H)
  const size_t A_OFF = C_OFF + (size_t)BB * HH;        // attn (T,B,S)

  // workspace carve-up (~96 MB)
  char* base = (char*)d_ws;
  size_t off = 0;
  auto carve = [&](size_t bytes) -> char* {
    char* r = base + off;
    off = (off + bytes + 255) & ~(size_t)255;
    return r;
  };
  bf16*  outW_bf = (bf16*)carve((size_t)VV * HH * 2);
  bf16*  Wih_bf  = (bf16*)carve((size_t)4 * HH * 2 * HH * 2);
  bf16*  Whh_bf  = (bf16*)carve((size_t)4 * HH * HH * 2);
  bf16*  W1_bf   = (bf16*)carve((size_t)HH * HH * 2);
  float* keys    = (float*)carve((size_t)SS * HH * 4);
  float* hst     = (float*)carve((size_t)BB * HH * 4);
  float* cst     = (float*)carve((size_t)BB * HH * 4);
  bf16*  hbf     = (bf16*)carve((size_t)BB * HH * 2);
  float* qW1     = (float*)carve((size_t)BB * HH * 4);
  float* scores  = (float*)carve((size_t)BB * SS * 4);
  bf16*  xbf     = (bf16*)carve((size_t)BB * 2 * HH * 2);
  float* gates   = (float*)carve((size_t)BB * 4 * HH * 4);

  // one-time weight conversions (bf16 weights then live in L2 across all steps)
  k_cvt_bf16<<<4096, 256, 0, stream>>>(outW, outW_bf, (size_t)VV * HH);
  k_cvt_bf16<<<2048, 256, 0, stream>>>(W_ih, Wih_bf, (size_t)4 * HH * 2 * HH);
  k_cvt_bf16<<<1024, 256, 0, stream>>>(W_hh, Whh_bf, (size_t)4 * HH * HH);
  k_cvt_bf16<<<512, 256, 0, stream>>>(W1, W1_bf, (size_t)HH * HH);

  // bridge + keys (one-shot, f32)
  k_matmul_nt_f32<<<(BB * HH + 255) / 256, 256, 0, stream>>>(enc_h, br1W, br1b, hst, BB, HH, HH);
  k_matmul_nt_f32<<<(BB * HH + 255) / 256, 256, 0, stream>>>(enc_c, br2W, br2b, cst, BB, HH, HH);
  k_cvt_bf16<<<256, 256, 0, stream>>>(hst, hbf, (size_t)BB * HH);
  k_matmul_nt_f32<<<(SS * HH + 255) / 256, 256, 0, stream>>>(enc, W2, b2, keys, SS, HH, HH);

  for (int t = 0; t < TT; ++t) {
    // q @ W1^T + b1   (WMMA)
    k_gemm_bf16_nt_m64<<<HH / 128, 256, 0, stream>>>(
        hbf, W1_bf, b1, qW1, HH, HH, (long long)HH, 0);
    // energies + softmax + context
    k_attn_scores<<<dim3(SS, BB), 256, 0, stream>>>(qW1, keys, Vw, Vb, scores);
    k_softmax_ctx<<<BB, 256, 0, stream>>>(scores, enc, out + A_OFF, xbf, t);
    k_pack_emb<<<BB, 256, 0, stream>>>(emb, tgt, xbf, t);
    // gates = x @ W_ih^T + b_ih + q @ W_hh^T + b_hh   (two WMMA GEMMs)
    k_gemm_bf16_nt_m64<<<4 * HH / 128, 256, 0, stream>>>(
        xbf, Wih_bf, b_ih, gates, 4 * HH, 2 * HH, (long long)4 * HH, 0);
    k_gemm_bf16_nt_m64<<<4 * HH / 128, 256, 0, stream>>>(
        hbf, Whh_bf, b_hh, gates, 4 * HH, HH, (long long)4 * HH, 1);
    // LSTM cell -> h,c state (+ bf16 h for next GEMMs, + final outputs)
    k_lstm_cell<<<BB, 256, 0, stream>>>(gates, hst, cst, hbf, out + H_OFF, out + C_OFF);
    // logits straight into final d_out slot, row stride T*V   (dominant WMMA GEMM)
    k_gemm_bf16_nt_m64<<<VV / 128, 256, 0, stream>>>(
        hbf, outW_bf, outb, out + (size_t)t * VV, VV, HH, (long long)TT * VV, 0);
    // in-place log_softmax over V
    k_log_softmax<<<BB, 256, 0, stream>>>(out, t);
  }
}
